// Variance_Predictor_Block_86165633892605
// MI455X (gfx1250) — matrix operand order repaired and checked by reading, not compiled, not run on hardware
//
#include <hip/hip_runtime.h>
#include <hip/hip_bf16.h>

// ---------------------------------------------------------------------------
// Variance-predictor block for MI455X (gfx1250, wave32, WMMA).
// The 6 big convs (256->256, k=3) run as bf16 WMMA GEMMs (f32 accumulate):
//   M=256 (Cout), K=768 (Cin*3), N=8192 (B*T)  => 19.3 GFLOP total,
// which at WMMA bf16 rates is ~memory-roofline (~220 MB of traffic @23.3TB/s).
// BatchNorm (training-mode batch stats) is folded as a per-channel affine
// into the next conv's LDS staging; bias+Mish fused into the GEMM epilogue.
//
// Input flattening assumption (setup_inputs() insertion order, nested dicts
// flattened in insertion order):
//   0 encodings[16,256,512] 1 means_p 2 log_std_p 3 encoding_masks[16,1,512]
//   4 durations[16,512] i32  5 log_f0s[16,512]  6 energies[16,512]
//   7..16  dur: w0,b0,g0,be0,w1,b1,g1,be1,wp,bp
//   17..26 f0 : same order        27..36 en : same order
//   37 f0_emb_w[256,1,9] 38 f0_emb_b 39 en_emb_w 40 en_emb_b 41 T_dec
// ---------------------------------------------------------------------------

typedef __attribute__((ext_vector_type(16))) __bf16 v16bf;
typedef __attribute__((ext_vector_type(8)))  float  v8f;

#define BATCH 16
#define DCH   256
#define TENC  512
#define TDEC  4096
#define LROW  272   // LDS row stride in bf16 elems (mult of 16 -> 32B aligned)

__device__ __forceinline__ unsigned short f2bf(float x) {
  union { float f; unsigned u; } v; v.f = x;
  unsigned r = v.u + 0x7FFFu + ((v.u >> 16) & 1u);   // round-to-nearest-even
  return (unsigned short)(r >> 16);
}

__device__ __forceinline__ float mishf(float x) {
  float sp = (x > 20.f) ? x : log1pf(__expf(x));     // softplus (stable)
  return x * tanhf(sp);
}

// --- pack f32 conv weight [256][256][3] into WMMA A-fragment bf16 layout ----
// packed layout: [m_tile(16)][cin_chunk(8)][kk(3)][lane(32)][elem(16)]
__global__ void pack_w_kernel(const float* __restrict__ w,
                              unsigned short* __restrict__ pw) {
  int e    = blockIdx.x * 256 + threadIdx.x;   // 0..196607
  int elem = e & 15;
  int lane = (e >> 4) & 31;
  int kk   = (e >> 9) % 3;
  int cc   = (e / 1536) & 7;
  int mt   = e / 12288;
  int co   = mt * 16 + (lane & 15);
  // 16-bit A 16x32 layout (contiguous, matches B): lane<16 holds K 0-15; lane>=16: 16-31
  int kloc = ((lane < 16) ? 0 : 16) + elem;
  int ci   = cc * 32 + kloc;
  pw[e] = f2bf(w[(co * DCH + ci) * 3 + kk]);
}

// --- h = mish(conv1d(affine(x), Wpacked, bias)), x:[B,256,512] f32 ----------
__global__ void __launch_bounds__(256)
conv_gemm_mish_kernel(const float* __restrict__ x,
                      const unsigned short* __restrict__ pw,
                      const float* __restrict__ bias,
                      const float* __restrict__ scale,   // null => identity
                      const float* __restrict__ shift,
                      float* __restrict__ out) {
  __shared__ __align__(32) unsigned short ldsx[66 * LROW];
  const int b   = blockIdx.z;
  const int t0  = blockIdx.x * 64;    // T tile
  const int c0  = blockIdx.y * 128;   // Cout tile
  const int tid = threadIdx.x;

  // Stage x[b, 0:256, t0-1 : t0+65] -> LDS[t][ci] as bf16, affine folded in.
  const float* xb = x + (size_t)b * DCH * TENC;
  for (int e = tid; e < 66 * DCH; e += 256) {
    int trow = e % 66;
    int ci   = e / 66;
    int tg   = t0 - 1 + trow;
    float v  = 0.f;
    if (tg >= 0 && tg < TENC) v = xb[ci * TENC + tg];
    if (scale) v = v * scale[ci] + shift[ci];
    ldsx[trow * LROW + ci] = f2bf(v);
  }
  __syncthreads();

  const int lane = tid & 31;
  const int wid  = tid >> 5;
  const int wm   = wid & 3;   // cout offset 32*wm
  const int wn   = wid >> 2;  // t    offset 32*wn

  v8f acc00 = {}, acc01 = {}, acc10 = {}, acc11 = {};
  const int mt0 = (c0 >> 4) + wm * 2;
  const int nl  = lane & 15;
  const int kb  = (lane < 16) ? 0 : 16;   // 16-bit B 32x16: lane<16 K 0-15

  for (int cc = 0; cc < 8; ++cc) {
#pragma unroll
    for (int kk = 0; kk < 3; ++kk) {
      v16bf a0 = *(const v16bf*)(pw + ((((mt0 + 0) * 8 + cc) * 3 + kk) * 32 + lane) * 16);
      v16bf a1 = *(const v16bf*)(pw + ((((mt0 + 1) * 8 + cc) * 3 + kk) * 32 + lane) * 16);
      v16bf b0 = *(const v16bf*)&ldsx[(wn * 32 +  0 + nl + kk) * LROW + cc * 32 + kb];
      v16bf b1 = *(const v16bf*)&ldsx[(wn * 32 + 16 + nl + kk) * LROW + cc * 32 + kb];
      acc00 = __builtin_amdgcn_wmma_f32_16x16x32_bf16(false, a0, false, b0, (short)0, acc00, false, false);
      acc01 = __builtin_amdgcn_wmma_f32_16x16x32_bf16(false, a0, false, b1, (short)0, acc01, false, false);
      acc10 = __builtin_amdgcn_wmma_f32_16x16x32_bf16(false, a1, false, b0, (short)0, acc10, false, false);
      acc11 = __builtin_amdgcn_wmma_f32_16x16x32_bf16(false, a1, false, b1, (short)0, acc11, false, false);
    }
  }

  // Epilogue: C/D 16x16 f32 layout: lane = N col; VGPR r -> M = r (+8 if lane>=16)
  float* ob = out + (size_t)b * DCH * TENC;
  const int mofs = (lane < 16) ? 0 : 8;
  v8f accs[2][2] = {{acc00, acc01}, {acc10, acc11}};
#pragma unroll
  for (int mi = 0; mi < 2; ++mi)
#pragma unroll
    for (int ni = 0; ni < 2; ++ni) {
      int tcol = t0 + wn * 32 + ni * 16 + nl;
#pragma unroll
      for (int r = 0; r < 8; ++r) {
        int co = c0 + wm * 32 + mi * 16 + mofs + r;
        ob[co * TENC + tcol] = mishf(accs[mi][ni][r] + bias[co]);
      }
    }
}

// --- batch-stats over (B,T) per channel -> folded affine scale/shift --------
__global__ void bnstats_kernel(const float* __restrict__ h,
                               const float* __restrict__ g,
                               const float* __restrict__ be,
                               float* __restrict__ scale,
                               float* __restrict__ shift) {
  __shared__ float s1[256], s2[256];
  int c = blockIdx.x, tid = threadIdx.x;
  float a = 0.f, q = 0.f;
  for (int i = tid; i < BATCH * TENC; i += 256) {
    int bb = i >> 9, tt = i & 511;
    float v = h[((size_t)bb * DCH + c) * TENC + tt];
    a += v; q += v * v;
  }
  s1[tid] = a; s2[tid] = q; __syncthreads();
  for (int off = 128; off > 0; off >>= 1) {
    if (tid < off) { s1[tid] += s1[tid + off]; s2[tid] += s2[tid + off]; }
    __syncthreads();
  }
  if (tid == 0) {
    float inv  = 1.f / (BATCH * TENC);
    float m    = s1[0] * inv;
    float var  = s2[0] * inv - m * m;
    float rstd = rsqrtf(var + 1e-5f);
    float sc   = g[c] * rstd;
    scale[c] = sc; shift[c] = be[c] - m * sc;
  }
}

// --- 1x1 projection over channels (+optional mask fill) ---------------------
__global__ void proj_kernel(const float* __restrict__ h,
                            const float* __restrict__ scale,
                            const float* __restrict__ shift,
                            const float* __restrict__ wp,
                            const float* __restrict__ bp,
                            const float* __restrict__ msk,
                            float fill,
                            float* __restrict__ out) {
  int g = blockIdx.x * 256 + threadIdx.x;  // B*T = 8192
  int b = g >> 9, t = g & 511;
  float acc = 0.f, base = bp[0];
  for (int c = 0; c < DCH; ++c) {
    float w = wp[c];
    base += shift[c] * w;
    acc  += h[((size_t)b * DCH + c) * TENC + t] * (scale[c] * w);
  }
  float o = base + acc;
  if (msk) o = (msk[b * TENC + t] > 0.f) ? o : fill;
  out[g] = o;
}

// --- means = means_p + (f0_emb + en_emb) * mask -----------------------------
__global__ void embed_means_kernel(const float* __restrict__ means_p,
                                   const float* __restrict__ msk,
                                   const float* __restrict__ lf0,
                                   const float* __restrict__ en,
                                   const float* __restrict__ f0w,
                                   const float* __restrict__ f0b,
                                   const float* __restrict__ enw,
                                   const float* __restrict__ enb,
                                   float* __restrict__ means) {
  int t = blockIdx.x * 256 + threadIdx.x;
  int c = blockIdx.y, b = blockIdx.z;
  float a = f0b[c], e = enb[c];
#pragma unroll
  for (int k = 0; k < 9; ++k) {
    int tt = t + k - 4;
    float v0 = 0.f, v1 = 0.f;
    if (tt >= 0 && tt < TENC) { v0 = lf0[b * TENC + tt]; v1 = en[b * TENC + tt]; }
    a += f0w[c * 9 + k] * v0;
    e += enw[c * 9 + k] * v1;
  }
  float m = msk[b * TENC + t];
  size_t o = ((size_t)b * DCH + c) * TENC + t;
  means[o] = means_p[o] + (a + e) * m;
}

// --- cumsum(durations) + searchsorted(right) -> idx[b, T_dec] ---------------
__global__ void regulate_idx_kernel(const int* __restrict__ dur,
                                    int* __restrict__ idx) {
  __shared__ int cum[512];
  int b = blockIdx.x, tid = threadIdx.x;
  cum[tid] = dur[b * TENC + tid];
  __syncthreads();
  for (int off = 1; off < 512; off <<= 1) {
    int v = (tid >= off) ? cum[tid - off] : 0;
    __syncthreads();
    cum[tid] += v;
    __syncthreads();
  }
  for (int p = tid; p < TDEC; p += 512) {
    int lo = 0, hi = 512;               // first j with cum[j] > p (else 512)
    while (lo < hi) { int mid = (lo + hi) >> 1; if (cum[mid] <= p) lo = mid + 1; else hi = mid; }
    idx[b * TDEC + p] = lo;
  }
}

// --- length-regulate gather (idx==512 selects appended zero frame) ----------
__global__ void gather_kernel(const float* __restrict__ means,
                              const float* __restrict__ lstd,
                              const int* __restrict__ idx,
                              float* __restrict__ o1,
                              float* __restrict__ o2) {
  size_t g = (size_t)blockIdx.x * 256 + threadIdx.x;  // 16*256*4096
  int pos = (int)(g & 4095);
  int c   = (int)((g >> 12) & 255);
  int b   = (int)(g >> 20);
  int j   = idx[b * TDEC + pos];
  float v1 = 0.f, v2 = 0.f;
  if (j < TENC) {
    size_t o = ((size_t)b * DCH + c) * TENC + j;
    v1 = means[o]; v2 = lstd[o];
  }
  o1[g] = v1; o2[g] = v2;
}

extern "C" void kernel_launch(void* const* d_in, const int* in_sizes, int n_in,
                              void* d_out, int out_size, void* d_ws, size_t ws_size,
                              hipStream_t stream) {
  const float* enc     = (const float*)d_in[0];
  const float* means_p = (const float*)d_in[1];
  const float* lstd    = (const float*)d_in[2];
  const float* msk     = (const float*)d_in[3];
  const int*   dur     = (const int*)d_in[4];
  const float* lf0     = (const float*)d_in[5];
  const float* ener    = (const float*)d_in[6];
  const float* P[30];
  for (int i = 0; i < 30; ++i) P[i] = (const float*)d_in[7 + i];
  const float* f0ew = (const float*)d_in[37];
  const float* f0eb = (const float*)d_in[38];
  const float* enew = (const float*)d_in[39];
  const float* eneb = (const float*)d_in[40];

  // workspace carving (~28 MB)
  char* ws = (char*)d_ws;
  size_t off = 0;
  auto alloc = [&](size_t bytes) { void* p = ws + off; off = (off + bytes + 255) & ~(size_t)255; return p; };
  unsigned short* pwbuf[6];
  for (int i = 0; i < 6; ++i) pwbuf[i] = (unsigned short*)alloc((size_t)DCH * DCH * 3 * 2);
  float* h0 = (float*)alloc((size_t)BATCH * DCH * TENC * 4);
  float* h1 = (float*)alloc((size_t)BATCH * DCH * TENC * 4);
  float* mb = (float*)alloc((size_t)BATCH * DCH * TENC * 4);
  float* sc0 = (float*)alloc(256 * 4); float* sh0 = (float*)alloc(256 * 4);
  float* sc1 = (float*)alloc(256 * 4); float* sh1 = (float*)alloc(256 * 4);
  int* idxbuf = (int*)alloc((size_t)BATCH * TDEC * 4);

  float* out = (float*)d_out;
  const size_t OSZ = (size_t)BATCH * DCH * TDEC;
  float* o_dur = out + 2 * OSZ;
  float* o_f0  = o_dur + BATCH * TENC;
  float* o_en  = o_f0 + BATCH * TENC;

  // pack the 6 big conv weights to WMMA bf16 fragment layout
  const float* wbig[6] = { P[0], P[4], P[10], P[14], P[20], P[24] };
  for (int i = 0; i < 6; ++i)
    pack_w_kernel<<<768, 256, 0, stream>>>(wbig[i], pwbuf[i]);

  dim3 gg(TENC / 64, DCH / 128, BATCH);
  for (int p = 0; p < 3; ++p) {
    const float* b0  = P[p * 10 + 1];
    const float* g0  = P[p * 10 + 2];
    const float* be0 = P[p * 10 + 3];
    const float* b1  = P[p * 10 + 5];
    const float* g1  = P[p * 10 + 6];
    const float* be1 = P[p * 10 + 7];
    const float* wp  = P[p * 10 + 8];
    const float* bp  = P[p * 10 + 9];
    conv_gemm_mish_kernel<<<gg, 256, 0, stream>>>(enc, pwbuf[p * 2 + 0], b0, nullptr, nullptr, h0);
    bnstats_kernel<<<256, 256, 0, stream>>>(h0, g0, be0, sc0, sh0);
    conv_gemm_mish_kernel<<<gg, 256, 0, stream>>>(h0, pwbuf[p * 2 + 1], b1, sc0, sh0, h1);
    bnstats_kernel<<<256, 256, 0, stream>>>(h1, g1, be1, sc1, sh1);
    float* tgt = (p == 0) ? o_dur : (p == 1) ? o_f0 : o_en;
    const float* m = (p == 0) ? nullptr : msk;
    float fill = (p == 1) ? -5.0f : -1.5f;
    proj_kernel<<<32, 256, 0, stream>>>(h1, sc1, sh1, wp, bp, m, fill, tgt);
  }

  embed_means_kernel<<<dim3(2, DCH, BATCH), 256, 0, stream>>>(
      means_p, msk, lf0, ener, f0ew, f0eb, enew, eneb, mb);
  regulate_idx_kernel<<<BATCH, 512, 0, stream>>>(dur, idxbuf);
  gather_kernel<<<65536, 256, 0, stream>>>(mb, lstd, idxbuf, out, out + OSZ);
}
